// RadixMajorNaiveImp_5669356831566
// MI455X (gfx1250) — matrix operand order repaired and checked by reading, not compiled, not run on hardware
//
#include <hip/hip_runtime.h>
#include <hip/hip_bf16.h>
#include <stdint.h>

// ---------------- problem constants ----------------
#define BSZ   32
#define CIN_  256
#define HH    56
#define WW    56
#define PIX   (HH*WW)        // 3136
#define CH_   256
#define RR    2
#define KK    2
#define GW_   128
#define IC_   64
#define OC_   (CH_*RR)       // 512 conv output channels
#define ICG   64             // input channels per conv group (CIN/(K*R))
#define KDIM  576            // ICG * 3 * 3  (GEMM K)
#define ASTR  584            // LDS row stride (bf16 elems), padded for bank spread

typedef __attribute__((ext_vector_type(16))) __bf16       v16bf;
typedef __attribute__((ext_vector_type(8)))  float        v8f;
typedef __attribute__((ext_vector_type(4)))  unsigned int v4u;
typedef __attribute__((ext_vector_type(2)))  unsigned int v2u;
typedef __attribute__((ext_vector_type(4)))  float        v4f;

static __device__ __forceinline__ unsigned short f2bf(float f) {
    unsigned u = __float_as_uint(f);
    unsigned r = u + 0x7FFFu + ((u >> 16) & 1u);   // round-to-nearest-even
    return (unsigned short)(r >> 16);
}
static __device__ __forceinline__ float bf2f(unsigned short h) {
    return __uint_as_float(((unsigned)h) << 16);
}

// ---------------- kernel 1: conv weights f32 -> bf16 (packed pairs) ----------
__global__ __launch_bounds__(256) void cvt_weights(const float* __restrict__ w,
                                                   unsigned int* __restrict__ o,
                                                   int npairs) {
    int i = blockIdx.x * 256 + threadIdx.x;
    if (i >= npairs) return;
    unsigned lo = f2bf(w[2*i]);
    unsigned hi = f2bf(w[2*i+1]);
    o[i] = lo | (hi << 16);
}

// ---------------- kernel 2: implicit-GEMM grouped conv, bf16 WMMA ------------
// grid = (49 pixel-tiles, B, 4 conv groups), block = 256 (8 waves)
// wave (wM = wave&3, wN = wave>>2): 16 pixels x 64 channels = 4 WMMA N-tiles
__global__ __launch_bounds__(256) void conv_wmma(
    const float* __restrict__ x, const unsigned short* __restrict__ wbf,
    const float* __restrict__ conv_b, unsigned short* __restrict__ ybf,
    float* __restrict__ gapsum)
{
    __shared__ __align__(16) unsigned short Alds[64 * ASTR];  // 74,752 B
    __shared__ float gapLds[GW_];

    const int tile = blockIdx.x;            // 0..48 (49*64 = 3136 pixels)
    const int b    = blockIdx.y;
    const int g    = blockIdx.z;            // conv group = r*K + k
    const int tid  = threadIdx.x;
    const int pwg  = tile * 64;             // first pixel of this tile

    if (tid < GW_) gapLds[tid] = 0.f;

    // ---- im2col of 64 pixels x 576 patch-elems into LDS (bf16) ----
    for (int idx = tid; idx < 64 * KDIM; idx += 256) {
        int m  = idx / KDIM;
        int kk = idx - m * KDIM;            // kk = ic*9 + kh*3 + kw (matches OIHW)
        int ic = kk / 9;
        int t9 = kk - ic * 9;
        int kh = t9 / 3, kw = t9 - kh * 3;
        int p  = pwg + m;
        int h  = p / WW, w = p - h * WW;
        int hi = h + kh - 1, wi = w + kw - 1;
        float val = 0.f;
        if ((unsigned)hi < (unsigned)HH && (unsigned)wi < (unsigned)WW)
            val = x[((size_t)(b * CIN_ + g * ICG + ic)) * PIX + hi * WW + wi];
        Alds[m * ASTR + kk] = f2bf(val);
    }
    __syncthreads();

    const int wave = tid >> 5;
    const int lane = tid & 31;
    const int wM   = wave & 3;              // pixel sub-tile
    const int wN   = wave >> 2;             // channel half (0/1)
    const int lhi  = lane >> 4;             // half-wave selector
    const int llo  = lane & 15;

    v8f acc[4] = {};
    const unsigned short* aBase = &Alds[(wM * 16 + llo) * ASTR];

    for (int ks = 0; ks < KDIM; ks += 32) {
        // A frag: lanes 0-15 -> K{ks..ks+7, ks+16..ks+23}; lanes 16-31 -> +8
        union { v4u u[2]; v16bf bf; } af;
        const int ak = ks + lhi * 8;
        af.u[0] = *(const v4u*)(aBase + ak);
        af.u[1] = *(const v4u*)(aBase + ak + 16);
        #pragma unroll
        for (int nt = 0; nt < 4; ++nt) {
            // B frag: lane col n = llo, lanes 0-15 hold K{ks..ks+7, ks+16..ks+23}; lanes 16-31 -> +8
            const int oc = g * 128 + wN * 64 + nt * 16 + llo;
            const unsigned short* bp = wbf + (size_t)oc * KDIM + ks + lhi * 8;
            union { v4u u[2]; v16bf bf; } bf_;
            bf_.u[0] = *(const v4u*)(bp);
            bf_.u[1] = *(const v4u*)(bp + 16);
            acc[nt] = __builtin_amdgcn_wmma_f32_16x16x32_bf16(
                false, af.bf, false, bf_.bf, (short)0, acc[nt], false, false);
        }
    }

    // ---- epilogue: bias + ReLU, pack bf16, 16B store, GAP partial sums ----
    const int prow = pwg + wM * 16 + lhi * 8;   // C layout: lane holds M = v + 8*lhi
    #pragma unroll
    for (int nt = 0; nt < 4; ++nt) {
        const int ocl  = wN * 64 + nt * 16 + llo;   // = gw (channel within group)
        const int oc   = g * 128 + ocl;
        const float bias = conv_b[oc];
        unsigned short hv[8];
        float s = 0.f;
        #pragma unroll
        for (int v = 0; v < 8; ++v) {
            float yv = acc[nt][v] + bias;
            yv = yv > 0.f ? yv : 0.f;
            s += yv;
            hv[v] = f2bf(yv);
        }
        v4u st;
        #pragma unroll
        for (int i = 0; i < 4; ++i)
            st[i] = (unsigned)hv[2*i] | ((unsigned)hv[2*i+1] << 16);
        *(v4u*)(ybf + ((size_t)(b * OC_ + oc)) * PIX + prow) = st;  // 8 px contiguous
        atomicAdd(&gapLds[ocl], s);
    }
    __syncthreads();
    if (tid < GW_)   // sum over radix merges here: g=k and g=2+k share (b,k,gw)
        atomicAdd(&gapsum[(b * KK + (g & 1)) * GW_ + tid], gapLds[tid]);
}

// ---------------- kernel 3: GAP mean -> FC1 -> FC2 -> softmax over radix -----
__global__ __launch_bounds__(256) void attention_fc(
    const float* __restrict__ gapsum, const float* __restrict__ w1,
    const float* __restrict__ b1, const float* __restrict__ w2,
    const float* __restrict__ b2, float* __restrict__ att)
{
    __shared__ float gbuf[GW_];
    __shared__ float hbuf[IC_];
    __shared__ float abuf[RR * GW_];
    const int bk = blockIdx.x;                 // 0..63
    const int b = bk >> 1, k = bk & 1;
    const int tid = threadIdx.x;

    if (tid < GW_) gbuf[tid] = gapsum[(b * KK + k) * GW_ + tid] * (1.f / PIX);
    __syncthreads();
    if (tid < IC_) {
        float h = b1[k * IC_ + tid];
        for (int gw = 0; gw < GW_; ++gw)
            h += gbuf[gw] * w1[(k * GW_ + gw) * IC_ + tid];
        hbuf[tid] = h;
    }
    __syncthreads();
    {
        float a = b2[k * (RR * GW_) + tid];
        for (int i = 0; i < IC_; ++i)
            a += hbuf[i] * w2[(k * IC_ + i) * (RR * GW_) + tid];
        abuf[tid] = a;
    }
    __syncthreads();
    if (tid < GW_) {
        float a0 = abuf[tid], a1 = abuf[GW_ + tid];
        float mx = fmaxf(a0, a1);
        float e0 = __expf(a0 - mx), e1 = __expf(a1 - mx);
        float inv = 1.f / (e0 + e1);
        att[((b * KK + k) * RR + 0) * GW_ + tid] = e0 * inv;
        att[((b * KK + k) * RR + 1) * GW_ + tid] = e1 * inv;
    }
}

// ---------------- kernel 4: out = att_r0*y_r0 + att_r1*y_r1 (stream) ---------
__global__ __launch_bounds__(256) void apply_att(
    const unsigned short* __restrict__ ybf, const float* __restrict__ att,
    float* __restrict__ out)
{
    const long long total4 = (long long)BSZ * CH_ * PIX / 4;
    long long idx = (long long)blockIdx.x * 256 + threadIdx.x;
    if (idx >= total4) return;
    int p4 = (int)(idx % (PIX / 4));
    long long rest = idx / (PIX / 4);
    int c = (int)(rest % CH_);
    int b = (int)(rest / CH_);
    int k = c >> 7, gw = c & 127;
    int p = p4 * 4;
    float a0 = att[((b * KK + k) * RR + 0) * GW_ + gw];
    float a1 = att[((b * KK + k) * RR + 1) * GW_ + gw];
    size_t y0 = ((size_t)(b * OC_ +        k * GW_ + gw)) * PIX + p;   // r=0
    size_t y1 = ((size_t)(b * OC_ + 256 +  k * GW_ + gw)) * PIX + p;   // r=1
    v2u u0 = *(const v2u*)(ybf + y0);
    v2u u1 = *(const v2u*)(ybf + y1);
    v4f o;
    #pragma unroll
    for (int j = 0; j < 4; ++j) {
        unsigned short h0 = (unsigned short)((u0[j >> 1] >> ((j & 1) * 16)) & 0xFFFFu);
        unsigned short h1 = (unsigned short)((u1[j >> 1] >> ((j & 1) * 16)) & 0xFFFFu);
        o[j] = a0 * bf2f(h0) + a1 * bf2f(h1);
    }
    *(v4f*)(out + ((size_t)(b * CH_ + c)) * PIX + p) = o;
}

// ---------------- host ----------------
extern "C" void kernel_launch(void* const* d_in, const int* in_sizes, int n_in,
                              void* d_out, int out_size, void* d_ws, size_t ws_size,
                              hipStream_t stream) {
    (void)in_sizes; (void)n_in; (void)out_size; (void)ws_size;
    const float* x      = (const float*)d_in[0];
    const float* conv_w = (const float*)d_in[1];
    const float* conv_b = (const float*)d_in[2];
    const float* w1     = (const float*)d_in[3];
    const float* b1     = (const float*)d_in[4];
    const float* w2     = (const float*)d_in[5];
    const float* b2     = (const float*)d_in[6];
    float* out          = (float*)d_out;

    char* ws = (char*)d_ws;
    unsigned short* wbf   = (unsigned short*)(ws);                      // 589,824 B
    float*          gapsv = (float*)(ws + (1 << 20));                   // 32 KB
    float*          attv  = (float*)(ws + (1 << 20) + (1 << 16));       // 64 KB
    unsigned short* ybf   = (unsigned short*)(ws + (2 << 20));          // ~102.8 MB

    hipMemsetAsync(gapsv, 0, (size_t)BSZ * KK * GW_ * sizeof(float), stream);

    const int npairs = OC_ * ICG * 9 / 2;   // 147456
    cvt_weights<<<(npairs + 255) / 256, 256, 0, stream>>>(conv_w, (unsigned int*)wbf, npairs);

    conv_wmma<<<dim3(49, BSZ, 4), 256, 0, stream>>>(x, wbf, conv_b, ybf, gapsv);

    attention_fc<<<BSZ * KK, 256, 0, stream>>>(gapsv, w1, b1, w2, b2, attv);

    const long long total4 = (long long)BSZ * CH_ * PIX / 4;
    apply_att<<<(int)((total4 + 255) / 256), 256, 0, stream>>>(ybf, attv, out);
}
